// EfficientAtt_13649406067216
// MI455X (gfx1250) — compile-verified, matrix operand order checked by reading, not run on hardware
//
#include <hip/hip_runtime.h>
#include <hip/hip_bf16.h>

typedef __attribute__((ext_vector_type(16))) _Float16 v16h;
typedef __attribute__((ext_vector_type(8)))  _Float16 v8h;
typedef __attribute__((ext_vector_type(8)))  float    v8f;

#define EPSV 1e-5f

__device__ __forceinline__ float hswish(float x) {
    float t = fminf(fmaxf(x + 3.0f, 0.0f), 6.0f);
    return x * t * (1.0f / 6.0f);
}

// Async global->LDS copy of 16 bytes per lane (GLOBAL_LOAD_ASYNC_TO_LDS_B128,
// GVS mode: SGPR64 base + per-lane signed 32-bit byte offset). Tracked by
// ASYNCcnt.
__device__ __forceinline__ void async_ld_b128(void* lds_dst, const void* gbase, int byte_off) {
    unsigned ldsa = (unsigned)(unsigned long long)lds_dst;  // low 32 bits = LDS offset
    asm volatile("global_load_async_to_lds_b128 %0, %1, %2"
                 :: "v"(ldsa), "v"(byte_off), "s"(gbase)
                 : "memory");
}
// Wait until ASYNCcnt <= N (N = async ops still allowed in flight, i.e. the
// prefetched next tile). N=0 drains everything.
template<int N>
__device__ __forceinline__ void wait_async() {
    asm volatile("s_wait_asynccnt %0" :: "i"(N) : "memory");
}

// Load a 16x32 f16 WMMA fragment (A-layout) from an LDS row-major tile.
// Element (m, k) lives at base[(row0+m)*ldl + k].
// lane<16: row m=lane, K in {0..7, 16..23}; lane>=16: row m=lane-16, K in {8..15, 24..31}.
__device__ __forceinline__ v16h ld_frag(const _Float16* base, int row0, int ldl) {
    int lane = threadIdx.x & 31;
    int m  = lane & 15;
    int kb = (lane >> 4) << 3;
    const _Float16* p = base + (row0 + m) * ldl;
    v16h f;
#pragma unroll
    for (int i = 0; i < 8; ++i) {
        int k = ((i >> 2) << 4) + kb + ((i & 3) << 1);
        f[2 * i]     = p[k];
        f[2 * i + 1] = p[k + 1];
    }
    return f;
}

// ---------------------------------------------------------------------------
// K0: pack weights to f16, transposed: WT[n][k] = W[k][n]
// ---------------------------------------------------------------------------
__global__ void __launch_bounds__(256) k_prep_weights(
    const float* __restrict__ wqkv, const float* __restrict__ wproj,
    _Float16* __restrict__ wqkvT, _Float16* __restrict__ wprojT)
{
    int idx = blockIdx.x * 256 + threadIdx.x;
    if (idx < 960 * 320) {
        int n = idx / 320, k = idx % 320;
        wqkvT[idx] = (_Float16)wqkv[k * 960 + n];
    }
    if (idx < 320 * 320) {
        int n = idx / 320, k = idx % 320;
        wprojT[idx] = (_Float16)wproj[k * 320 + n];
    }
}

// ---------------------------------------------------------------------------
// K1: QKV GEMM. C[32768,960] = x[32768,320] @ Wqkv[320,960] + b.
// Block tile 64x64, 8 waves (2 M-waves x 4 N-waves), each wave 32x16.
// Double-buffered staging: B tile via async-to-LDS (1 op/lane in flight),
// A (f32) tile via load+cvt+ds_store issued a step ahead.
// Epilogue writes channel-major f16: qkv_cm[(j*8+b)*320 + c][token]
// ---------------------------------------------------------------------------
__global__ void __launch_bounds__(256) k_gemm_qkv(
    const float* __restrict__ A, const _Float16* __restrict__ WT,
    const float* __restrict__ bias, _Float16* __restrict__ qkv_cm)
{
    __shared__ _Float16 la[2][64][48];
    __shared__ _Float16 lb[2][64][48];
    const int m0 = blockIdx.x * 64;
    const int n0 = blockIdx.y * 64;
    const int tid = threadIdx.x;
    const int wave = tid >> 5;
    const int wm = wave >> 2;   // 0..1
    const int wn = wave & 3;    // 0..3
    const int lane = tid & 31;
    const int r  = tid >> 2;          // staging row 0..63
    const int c0 = (tid & 3) << 3;    // staging col 0,8,16,24
    v8f acc0 = {}; v8f acc1 = {};
    // prologue: stage tile kt=0 into buffer 0
    async_ld_b128(&lb[0][r][c0], WT, ((n0 + r) * 320 + c0) * 2);
    {
        const float* ap = A + (m0 + r) * 320 + c0;
#pragma unroll
        for (int i = 0; i < 8; ++i) la[0][r][c0 + i] = (_Float16)ap[i];
    }
    int buf = 0;
    for (int kt = 0; kt < 320; kt += 32, buf ^= 1) {
        const bool nxt = (kt + 32) < 320;
        if (nxt) {
            // prefetch tile kt+32 into the other buffer while tile kt computes
            async_ld_b128(&lb[buf ^ 1][r][c0], WT, ((n0 + r) * 320 + kt + 32 + c0) * 2);
            const float* ap = A + (m0 + r) * 320 + kt + 32 + c0;
            __builtin_prefetch(ap, 0, 1);
#pragma unroll
            for (int i = 0; i < 8; ++i) la[buf ^ 1][r][c0 + i] = (_Float16)ap[i];
            wait_async<1>();   // tile kt's async done; kt+32's may stay in flight
        } else {
            wait_async<0>();
        }
        __syncthreads();
        v16h a0 = ld_frag(&la[buf][0][0], wm * 32 + 0,  48);
        v16h a1 = ld_frag(&la[buf][0][0], wm * 32 + 16, 48);
        v16h bf = ld_frag(&lb[buf][0][0], wn * 16,      48);
        acc0 = __builtin_amdgcn_wmma_f32_16x16x32_f16(false, a0, false, bf, (short)0, acc0, false, false);
        acc1 = __builtin_amdgcn_wmma_f32_16x16x32_f16(false, a1, false, bf, (short)0, acc1, false, false);
        __syncthreads();
    }
    const int cg = n0 + wn * 16 + (lane & 15);   // 0..959
    const int j  = cg / 320, c = cg % 320;
    const float bv = bias[cg];
#pragma unroll
    for (int sub = 0; sub < 2; ++sub) {
        v8f acc = sub ? acc1 : acc0;
        int mloc = wm * 32 + sub * 16 + ((lane >> 4) << 3);
        int mg = m0 + mloc;
        int b  = mg >> 12, nt = mg & 4095;
        v8h o;
#pragma unroll
        for (int q = 0; q < 8; ++q) o[q] = (_Float16)(acc[q] + bv);
        *(v8h*)(qkv_cm + ((((j * 8 + b) * 320 + c) << 12) + nt)) = o;
    }
}

// ---------------------------------------------------------------------------
// K2a: segment 0: xc[a][0..63][n] = hardswish(bn0(qkv[a][0..63][n]))
// ---------------------------------------------------------------------------
__global__ void __launch_bounds__(256) k_seg0(
    const _Float16* __restrict__ qkv_cm, _Float16* __restrict__ xc_cm,
    const float* __restrict__ g, const float* __restrict__ bb,
    const float* __restrict__ mm, const float* __restrict__ vv)
{
    int i = blockIdx.x * 256 + threadIdx.x;      // 24*64*4096
    int n = i & 4095, c = (i >> 12) & 63, a = i >> 18;
    float sc = g[c] * rsqrtf(vv[c] + EPSV);
    float sh = bb[c] - mm[c] * sc;
    float x = (float)qkv_cm[((a * 320 + c) << 12) + n] * sc + sh;
    xc_cm[((a * 256 + c) << 12) + n] = (_Float16)hswish(x);
}

// ---------------------------------------------------------------------------
// K2b: segments 1..3: depthwise kxk -> pointwise 64x64 -> BN -> hardswish
// Block: (pixel 8x8 tile, image a, seg). LDS stages the depthwise output.
// ---------------------------------------------------------------------------
__global__ void __launch_bounds__(256) k_sepconv(
    const _Float16* __restrict__ qkv_cm, _Float16* __restrict__ xc_cm,
    const float* dw1, const float* pw1, const float* dw2, const float* pw2,
    const float* dw3, const float* pw3,
    const float* g1, const float* b1, const float* m1, const float* v1,
    const float* g2, const float* b2, const float* m2, const float* v2,
    const float* g3, const float* b3, const float* m3, const float* v3)
{
    __shared__ float lds_dw[64][65];
    const int tile = blockIdx.x, a = blockIdx.y, z = blockIdx.z;
    const int seg = z + 1;
    const float *wd, *wp, *g, *bb, *mm, *vv; int kk;
    if (z == 0)      { wd = dw1; wp = pw1; g = g1; bb = b1; mm = m1; vv = v1; kk = 3; }
    else if (z == 1) { wd = dw2; wp = pw2; g = g2; bb = b2; mm = m2; vv = v2; kk = 5; }
    else             { wd = dw3; wp = pw3; g = g3; bb = b3; mm = m3; vv = v3; kk = 7; }
    const int py0 = (tile >> 3) << 3, px0 = (tile & 7) << 3;
    const int tid = threadIdx.x;
    const int hk = kk >> 1;
#pragma unroll
    for (int it = 0; it < 16; ++it) {
        int o = tid + (it << 8);
        int ci = o >> 6, p = o & 63;
        int py = py0 + (p >> 3), px = px0 + (p & 7);
        const _Float16* src = qkv_cm + ((a * 320 + seg * 64 + ci) << 12);
        const float* w = wd + ci * kk * kk;
        float s = 0.0f;
        for (int dy = 0; dy < kk; ++dy) {
            int iy = py + dy - hk;
            if (iy < 0 || iy > 63) continue;
            for (int dx = 0; dx < kk; ++dx) {
                int ix = px + dx - hk;
                if (ix < 0 || ix > 63) continue;
                s += (float)src[(iy << 6) + ix] * w[dy * kk + dx];
            }
        }
        lds_dw[ci][p] = s;
    }
    __syncthreads();
#pragma unroll
    for (int it = 0; it < 16; ++it) {
        int o = tid + (it << 8);
        int co = o >> 6, p = o & 63;
        const float* w = wp + co * 64;
        float s = 0.0f;
#pragma unroll 8
        for (int ci = 0; ci < 64; ++ci) s += lds_dw[ci][p] * w[ci];
        float sc = g[co] * rsqrtf(vv[co] + EPSV);
        float sh = bb[co] - mm[co] * sc;
        int pix = ((py0 + (p >> 3)) << 6) + px0 + (p & 7);
        xc_cm[((a * 256 + seg * 64 + co) << 12) + pix] = (_Float16)hswish(s * sc + sh);
    }
}

// ---------------------------------------------------------------------------
// K2c: Agg0 local branch: stack q/k/v of segment 4 (192 ch), dw 3x3,
// pw 192->64, LayerNorm over channels, hardswish -> xl_buf[b][n][64]
// ---------------------------------------------------------------------------
__global__ void __launch_bounds__(256) k_agg0(
    const _Float16* __restrict__ qkv_cm,
    const float* __restrict__ a0dw, const float* __restrict__ a0pw,
    const float* __restrict__ lng, const float* __restrict__ lnb,
    _Float16* __restrict__ xl_buf)
{
    __shared__ _Float16 lds_dw[192][72];
    __shared__ float    lds_pw[64][65];
    const int tile = blockIdx.x, b = blockIdx.y;
    const int py0 = (tile >> 3) << 3, px0 = (tile & 7) << 3;
    const int tid = threadIdx.x;
#pragma unroll
    for (int it = 0; it < 48; ++it) {
        int o = tid + (it << 8);
        int cc = o >> 6, p = o & 63;
        int j = cc / 64, ci = cc % 64;
        int py = py0 + (p >> 3), px = px0 + (p & 7);
        const _Float16* src = qkv_cm + (((j * 8 + b) * 320 + 256 + ci) << 12);
        const float* w = a0dw + cc * 9;
        float s = 0.0f;
        for (int dy = 0; dy < 3; ++dy) {
            int iy = py + dy - 1;
            if (iy < 0 || iy > 63) continue;
            for (int dx = 0; dx < 3; ++dx) {
                int ix = px + dx - 1;
                if (ix < 0 || ix > 63) continue;
                s += (float)src[(iy << 6) + ix] * w[dy * 3 + dx];
            }
        }
        lds_dw[cc][p] = (_Float16)s;
    }
    __syncthreads();
#pragma unroll
    for (int it = 0; it < 16; ++it) {
        int o = tid + (it << 8);
        int co = o >> 6, p = o & 63;
        const float* w = a0pw + co * 192;
        float s = 0.0f;
#pragma unroll 8
        for (int cc = 0; cc < 192; ++cc) s += (float)lds_dw[cc][p] * w[cc];
        lds_pw[p][co] = s;
    }
    __syncthreads();
    {
        int p = tid >> 2;
        int i0 = (tid & 3) << 4;
        float mu = 0.0f;
#pragma unroll 8
        for (int q = 0; q < 64; ++q) mu += lds_pw[p][q];
        mu *= (1.0f / 64.0f);
        float var = 0.0f;
#pragma unroll 8
        for (int q = 0; q < 64; ++q) { float d = lds_pw[p][q] - mu; var += d * d; }
        var *= (1.0f / 64.0f);
        float rs = rsqrtf(var + EPSV);
        int pix = ((py0 + (p >> 3)) << 6) + px0 + (p & 7);
        _Float16* dst = xl_buf + (((b << 12) + pix) << 6);
#pragma unroll
        for (int q = 0; q < 16; ++q) {
            int co = i0 + q;
            float val = (lds_pw[p][co] - mu) * rs * lng[co] + lnb[co];
            dst[co] = (_Float16)hswish(val);
        }
    }
}

// ---------------------------------------------------------------------------
// K3: softmax of k over tokens (N=4096), per (b, channel). Channel-major rows
// make the reduction fully contiguous.
// ---------------------------------------------------------------------------
__global__ void __launch_bounds__(256) k_softmax_k(
    const _Float16* __restrict__ xc_cm, _Float16* __restrict__ ksm)
{
    __shared__ float red[256];
    const int c = blockIdx.x, b = blockIdx.y;
    const _Float16* row = xc_cm + (((8 + b) * 256 + c) << 12);
    const int tid = threadIdx.x;
    float mx = -3.0e38f;
    for (int n = tid; n < 4096; n += 256) mx = fmaxf(mx, (float)row[n]);
    red[tid] = mx; __syncthreads();
    for (int s = 128; s > 0; s >>= 1) {
        if (tid < s) red[tid] = fmaxf(red[tid], red[tid + s]);
        __syncthreads();
    }
    mx = red[0]; __syncthreads();
    float sm = 0.0f;
    for (int n = tid; n < 4096; n += 256) sm += __expf((float)row[n] - mx);
    red[tid] = sm; __syncthreads();
    for (int s = 128; s > 0; s >>= 1) {
        if (tid < s) red[tid] += red[tid + s];
        __syncthreads();
    }
    float inv = 1.0f / red[0];
    _Float16* out = ksm + ((b * 256 + c) << 12);
    for (int n = tid; n < 4096; n += 256)
        out[n] = (_Float16)(__expf((float)row[n] - mx) * inv);
}

// ---------------------------------------------------------------------------
// K4: kv[b,h] = Ksm[32,4096] @ V[32,4096]^T via WMMA, K-loop over tokens.
// One block per (b,h): 4 waves, each owns one 16x16 tile of the 32x32 result.
// Double-buffered async-to-LDS staging (2 ops/lane in flight).
// ---------------------------------------------------------------------------
__global__ void __launch_bounds__(128) k_kv(
    const _Float16* __restrict__ ksm, const _Float16* __restrict__ xc_cm,
    float* __restrict__ kvbuf)
{
    __shared__ _Float16 lka[2][32][48];
    __shared__ _Float16 lvv[2][32][48];
    const int b = blockIdx.x >> 3, h = blockIdx.x & 7;
    const int tid = threadIdx.x;
    const int wave = tid >> 5, lane = tid & 31;
    const int wm = wave >> 1, wn = wave & 1;
    const _Float16* kb = ksm   + ((b * 256 + h * 32) << 12);
    const _Float16* vb = xc_cm + (((16 + b) * 256 + h * 32) << 12);
    const int r = tid >> 2, off = (tid & 3) << 3;
    v8f acc = {};
    // prologue: tile kt=0 into buffer 0
    {
        int g0 = ((r << 12) + off) * 2;
        async_ld_b128(&lka[0][r][off], kb, g0);
        async_ld_b128(&lvv[0][r][off], vb, g0);
    }
    int buf = 0;
    for (int kt = 0; kt < 4096; kt += 32, buf ^= 1) {
        const bool nxt = (kt + 32) < 4096;
        if (nxt) {
            int gn = ((r << 12) + kt + 32 + off) * 2;
            async_ld_b128(&lka[buf ^ 1][r][off], kb, gn);
            async_ld_b128(&lvv[buf ^ 1][r][off], vb, gn);
            wait_async<2>();   // current tile complete; next tile in flight
        } else {
            wait_async<0>();
        }
        __syncthreads();
        v16h a  = ld_frag(&lka[buf][0][0], wm * 16, 48);
        v16h bf = ld_frag(&lvv[buf][0][0], wn * 16, 48);
        acc = __builtin_amdgcn_wmma_f32_16x16x32_f16(false, a, false, bf, (short)0, acc, false, false);
        __syncthreads();
    }
    int col   = wn * 16 + (lane & 15);
    int rbase = wm * 16 + ((lane >> 4) << 3);
    float* dst = kvbuf + ((b * 8 + h) << 10);
#pragma unroll
    for (int q = 0; q < 8; ++q) dst[(rbase + q) * 32 + col] = acc[q];
}

// ---------------------------------------------------------------------------
// K5: CRPE depthwise conv on v (windows 3/5/7 by head group) + bias.
// ---------------------------------------------------------------------------
__global__ void __launch_bounds__(256) k_crpe(
    const _Float16* __restrict__ xc_cm,
    const float* __restrict__ d0, const float* __restrict__ bb0,
    const float* __restrict__ d1, const float* __restrict__ bb1,
    const float* __restrict__ d2, const float* __restrict__ bb2,
    _Float16* __restrict__ convv)
{
    const int c = blockIdx.x, b = blockIdx.y;
    int kk; const float* wd; float bv;
    if (c < 64)       { kk = 3; wd = d0 + c * 9;          bv = bb0[c]; }
    else if (c < 160) { kk = 5; wd = d1 + (c - 64) * 25;  bv = bb1[c - 64]; }
    else              { kk = 7; wd = d2 + (c - 160) * 49; bv = bb2[c - 160]; }
    const _Float16* src = xc_cm + (((16 + b) * 256 + c) << 12);
    _Float16* dst = convv + ((b * 256 + c) << 12);
    const int hk = kk >> 1;
    for (int p = threadIdx.x; p < 4096; p += 256) {
        int py = p >> 6, px = p & 63;
        float s = bv;
        for (int dy = 0; dy < kk; ++dy) {
            int iy = py + dy - hk;
            if (iy < 0 || iy > 63) continue;
            for (int dx = 0; dx < kk; ++dx) {
                int ix = px + dx - hk;
                if (ix < 0 || ix > 63) continue;
                s += (float)src[(iy << 6) + ix] * wd[dy * kk + dx];
            }
        }
        dst[p] = (_Float16)s;
    }
}

// ---------------------------------------------------------------------------
// K6: eff = q @ kv (single k-step WMMA), fused with crpe = q * conv_v.
// Block per (b, h, 64-token tile): 8 waves = 4 token-tiles x 2 vc-tiles.
// Writes y2 token-major f16.
// ---------------------------------------------------------------------------
__global__ void __launch_bounds__(256) k_eff_combine(
    const _Float16* __restrict__ xc_cm, const float* __restrict__ kvbuf,
    const _Float16* __restrict__ convv, _Float16* __restrict__ y2)
{
    __shared__ _Float16 lq[64][40];
    __shared__ _Float16 lkv[32][40];
    const int n0 = blockIdx.x * 64, h = blockIdx.y, b = blockIdx.z;
    const int tid = threadIdx.x;
    // stage q (transpose channel-major -> [token][channel]) coalesced over tokens
    {
        int c = tid >> 3, toff = (tid & 7) << 3;
        const _Float16* qs = xc_cm + ((b * 256 + h * 32 + c) << 12) + n0 + toff;
#pragma unroll
        for (int i = 0; i < 8; ++i) lq[toff + i][c] = qs[i];
    }
    // stage kv transposed: lkv[vc][c] = kv[c][vc], f32->f16
    {
        const float* kp = kvbuf + ((b * 8 + h) << 10);
#pragma unroll
        for (int i = 0; i < 4; ++i) {
            int e = tid * 4 + i;
            int c = e >> 5, vc = e & 31;
            lkv[vc][c] = (_Float16)kp[e];
        }
    }
    __syncthreads();
    const int wave = tid >> 5, lane = tid & 31;
    const int tm = wave >> 1, tn = wave & 1;
    v16h a  = ld_frag(&lq[0][0],  tm * 16, 40);
    v16h bf = ld_frag(&lkv[0][0], tn * 16, 40);
    v8f acc = {};
    acc = __builtin_amdgcn_wmma_f32_16x16x32_f16(false, a, false, bf, (short)0, acc, false, false);
    const float scale = 0.17677669529663688f;   // 32^-0.5
    const int vc = tn * 16 + (lane & 15);
    const _Float16* cv = convv + ((b * 256 + h * 32 + vc) << 12) + n0;
#pragma unroll
    for (int q = 0; q < 8; ++q) {
        int tl = tm * 16 + ((lane >> 4) << 3) + q;
        float qv = (float)lq[tl][vc];
        float val = scale * acc[q] + qv * (float)cv[tl];
        y2[(((b << 12) + n0 + tl) * 320) + h * 32 + vc] = (_Float16)val;
    }
}

// ---------------------------------------------------------------------------
// K6b: concat xl into y2[:, :, 256:320]
// ---------------------------------------------------------------------------
__global__ void __launch_bounds__(256) k_xlcopy(
    const _Float16* __restrict__ xl_buf, _Float16* __restrict__ y2)
{
    int i = blockIdx.x * 256 + threadIdx.x;   // 8*4096*64
    int c = i & 63, n = (i >> 6) & 4095, b = i >> 18;
    y2[(((b << 12) + n) * 320) + 256 + c] = xl_buf[i];
}

// ---------------------------------------------------------------------------
// K7: projection GEMM: out[32768,320] = y2[32768,320] @ Wproj[320,320] + b
// Both tiles f16 -> both async, double-buffered (2 ops/lane in flight).
// ---------------------------------------------------------------------------
__global__ void __launch_bounds__(256) k_gemm_proj(
    const _Float16* __restrict__ A, const _Float16* __restrict__ WT,
    const float* __restrict__ bias, float* __restrict__ out)
{
    __shared__ _Float16 la[2][64][48];
    __shared__ _Float16 lb[2][64][48];
    const int m0 = blockIdx.x * 64;
    const int n0 = blockIdx.y * 64;
    const int tid = threadIdx.x;
    const int wave = tid >> 5;
    const int wm = wave >> 2, wn = wave & 3;
    const int lane = tid & 31;
    const int r = tid >> 2, c0 = (tid & 3) << 3;
    v8f acc0 = {}; v8f acc1 = {};
    // prologue: tile kt=0 into buffer 0
    async_ld_b128(&la[0][r][c0], A,  ((m0 + r) * 320 + c0) * 2);
    async_ld_b128(&lb[0][r][c0], WT, ((n0 + r) * 320 + c0) * 2);
    int buf = 0;
    for (int kt = 0; kt < 320; kt += 32, buf ^= 1) {
        const bool nxt = (kt + 32) < 320;
        if (nxt) {
            async_ld_b128(&la[buf ^ 1][r][c0], A,  ((m0 + r) * 320 + kt + 32 + c0) * 2);
            async_ld_b128(&lb[buf ^ 1][r][c0], WT, ((n0 + r) * 320 + kt + 32 + c0) * 2);
            wait_async<2>();
        } else {
            wait_async<0>();
        }
        __syncthreads();
        v16h a0 = ld_frag(&la[buf][0][0], wm * 32 + 0,  48);
        v16h a1 = ld_frag(&la[buf][0][0], wm * 32 + 16, 48);
        v16h bf = ld_frag(&lb[buf][0][0], wn * 16,      48);
        acc0 = __builtin_amdgcn_wmma_f32_16x16x32_f16(false, a0, false, bf, (short)0, acc0, false, false);
        acc1 = __builtin_amdgcn_wmma_f32_16x16x32_f16(false, a1, false, bf, (short)0, acc1, false, false);
        __syncthreads();
    }
    const int ng = n0 + wn * 16 + (lane & 15);
    const float bv = bias[ng];
#pragma unroll
    for (int sub = 0; sub < 2; ++sub) {
        v8f acc = sub ? acc1 : acc0;
        int mg = m0 + wm * 32 + sub * 16 + ((lane >> 4) << 3);
#pragma unroll
        for (int q = 0; q < 8; ++q) out[(mg + q) * 320 + ng] = acc[q] + bv;
    }
}

// ---------------------------------------------------------------------------
// Launch
// ---------------------------------------------------------------------------
extern "C" void kernel_launch(void* const* d_in, const int* in_sizes, int n_in,
                              void* d_out, int out_size, void* d_ws, size_t ws_size,
                              hipStream_t stream) {
    (void)in_sizes; (void)n_in; (void)out_size; (void)ws_size;
    // inputs in setup_inputs() insertion order, params flattened in insertion
    // order (bn tuples flatten to 4 arrays each)
    const float* x        = (const float*)d_in[0];
    const float* w_qkv    = (const float*)d_in[1];
    const float* b_qkv    = (const float*)d_in[2];
    const float* dw1      = (const float*)d_in[3];
    const float* pw1      = (const float*)d_in[4];
    const float* dw2      = (const float*)d_in[5];
    const float* pw2      = (const float*)d_in[6];
    const float* dw3      = (const float*)d_in[7];
    const float* pw3      = (const float*)d_in[8];
    const float* bn0g = (const float*)d_in[9],  *bn0b = (const float*)d_in[10];
    const float* bn0m = (const float*)d_in[11], *bn0v = (const float*)d_in[12];
    const float* bn1g = (const float*)d_in[13], *bn1b = (const float*)d_in[14];
    const float* bn1m = (const float*)d_in[15], *bn1v = (const float*)d_in[16];
    const float* bn2g = (const float*)d_in[17], *bn2b = (const float*)d_in[18];
    const float* bn2m = (const float*)d_in[19], *bn2v = (const float*)d_in[20];
    const float* bn3g = (const float*)d_in[21], *bn3b = (const float*)d_in[22];
    const float* bn3m = (const float*)d_in[23], *bn3v = (const float*)d_in[24];
    const float* a0dw     = (const float*)d_in[25];
    const float* a0pw     = (const float*)d_in[26];
    const float* lng      = (const float*)d_in[27];
    const float* lnb      = (const float*)d_in[28];
    const float* cdw0     = (const float*)d_in[29];
    const float* cb0      = (const float*)d_in[30];
    const float* cdw1     = (const float*)d_in[31];
    const float* cb1      = (const float*)d_in[32];
    const float* cdw2     = (const float*)d_in[33];
    const float* cb2      = (const float*)d_in[34];
    const float* w_proj   = (const float*)d_in[35];
    const float* b_proj   = (const float*)d_in[36];
    float* out = (float*)d_out;

    // workspace layout (bytes). ksm/convv/y2 alias the qkv buffer, which is
    // dead once the aggregator kernels (seg0/sepconv/agg0) have run.
    char* ws = (char*)d_ws;
    _Float16* qkv_cm = (_Float16*)(ws);                     // 62,914,560 B  [dead after K2*]
    _Float16* ksm    = (_Float16*)(ws);                     // 16,777,216 B  (alias)
    _Float16* convv  = (_Float16*)(ws + 16777216);          // 16,777,216 B  (alias)
    _Float16* y2     = (_Float16*)(ws + 33554432);          // 20,971,520 B  (alias)
    _Float16* xc_cm  = (_Float16*)(ws + 62914560);          // 50,331,648 B
    float*    kvbuf  = (float*)   (ws + 113246208);         //    262,144 B
    _Float16* wqkvT  = (_Float16*)(ws + 113508352);         //    614,400 B
    _Float16* wprojT = (_Float16*)(ws + 114122752);         //    204,800 B
    _Float16* xl_buf = (_Float16*)(ws + 114327552);         //  4,194,304 B  (total ~118.5 MB)

    k_prep_weights<<<1200, 256, 0, stream>>>(w_qkv, w_proj, wqkvT, wprojT);
    k_gemm_qkv<<<dim3(512, 15), 256, 0, stream>>>(x, wqkvT, b_qkv, qkv_cm);
    k_seg0<<<24576, 256, 0, stream>>>(qkv_cm, xc_cm, bn0g, bn0b, bn0m, bn0v);
    k_sepconv<<<dim3(64, 24, 3), 256, 0, stream>>>(qkv_cm, xc_cm,
        dw1, pw1, dw2, pw2, dw3, pw3,
        bn1g, bn1b, bn1m, bn1v, bn2g, bn2b, bn2m, bn2v, bn3g, bn3b, bn3m, bn3v);
    k_agg0<<<dim3(64, 8), 256, 0, stream>>>(qkv_cm, a0dw, a0pw, lng, lnb, xl_buf);
    // qkv_cm dead from here; aliased buffers become live
    k_softmax_k<<<dim3(256, 8), 256, 0, stream>>>(xc_cm, ksm);
    k_kv<<<64, 128, 0, stream>>>(ksm, xc_cm, kvbuf);
    k_crpe<<<dim3(256, 8), 256, 0, stream>>>(xc_cm, cdw0, cb0, cdw1, cb1, cdw2, cb2, convv);
    k_eff_combine<<<dim3(64, 8, 8), 256, 0, stream>>>(xc_cm, kvbuf, convv, y2);
    k_xlcopy<<<8192, 256, 0, stream>>>(xl_buf, y2);
    k_gemm_proj<<<dim3(512, 5), 256, 0, stream>>>(y2, wprojT, b_proj, out);
}
